// MoELayer_7164005449798
// MI455X (gfx1250) — matrix operand order repaired and checked by reading, not compiled, not run on hardware
//
#include <hip/hip_runtime.h>
#include <hip/hip_bf16.h>
#include <math.h>

// ---------------- problem constants (match reference) ----------------
namespace {
constexpr int   kT   = 4096;   // B*S tokens
constexpr int   kH   = 1024;   // hidden
constexpr int   kI   = 2048;   // intermediate
constexpr int   kE   = 8;      // experts
constexpr float kAux = 0.01f;
constexpr float kZ   = 0.001f;
}

// ---------------- WMMA types ----------------
typedef __attribute__((ext_vector_type(16))) __bf16 v16bf;
typedef __attribute__((ext_vector_type(8)))  float  v8f;

union FragU {
    unsigned int u[8];
    v16bf        v;
};

// fp32 pair -> packed bf16x2 (header picks v_cvt_pk_bf16_f32 when available)
__device__ __forceinline__ unsigned int pack2_bf16(float lo, float hi) {
    union { __hip_bfloat162 h; unsigned int u; } cvt;
    cvt.h = __float22bfloat162_rn(float2{lo, hi});
    return cvt.u;
}

__device__ __forceinline__ unsigned short f2bf16(float f) {
    union { __hip_bfloat16 h; unsigned short u; } cvt;
    cvt.h = __float2bfloat16(f);
    return cvt.u;
}

// ---------------- zero init ----------------
__global__ void moe_zero_f_kernel(float* __restrict__ p, int n) {
    int i = blockIdx.x * blockDim.x + threadIdx.x;
    if (i < n) p[i] = 0.0f;
}
__global__ void moe_zero_i_kernel(int* __restrict__ p, int n) {
    int i = blockIdx.x * blockDim.x + threadIdx.x;
    if (i < n) p[i] = 0;
}

// ---------------- bulk fp32 -> bf16 weight conversion (optional path) --------
__global__ void moe_cvt_bf16_kernel(const float* __restrict__ src,
                                    unsigned short* __restrict__ dst,
                                    long n2)   // number of fp32 pairs
{
    long i = (long)blockIdx.x * blockDim.x + threadIdx.x;
    const long stride = (long)gridDim.x * blockDim.x;
    for (; i < n2; i += stride) {
        const float2 v = ((const float2*)src)[i];
        ((unsigned int*)dst)[i] = pack2_bf16(v.x, v.y);
    }
}

// ---------------- router: one wave32 per token ----------------
__global__ void moe_router_kernel(const float* __restrict__ x,      // [T,H]
                                  const float* __restrict__ gw,     // [E,H]
                                  int*   __restrict__ ecnt,         // [E]
                                  int*   __restrict__ elist,        // [E,T]
                                  float* __restrict__ ew,           // [E,T]
                                  float* __restrict__ zz)           // [T]
{
    const int tok  = (blockIdx.x * blockDim.x + threadIdx.x) >> 5;
    const int lane = threadIdx.x & 31;
    if (tok >= kT) return;

    const float* xr = x + (size_t)tok * kH;
    float xv[kH / 32];
#pragma unroll
    for (int j = 0; j < kH / 32; ++j) xv[j] = xr[lane + 32 * j];

    float logit[kE];
#pragma unroll
    for (int e = 0; e < kE; ++e) {
        const float* g = gw + (size_t)e * kH;
        float acc = 0.0f;
#pragma unroll
        for (int j = 0; j < kH / 32; ++j) acc += xv[j] * g[lane + 32 * j];
#pragma unroll
        for (int o = 16; o > 0; o >>= 1) acc += __shfl_xor(acc, o, 32);
        logit[e] = acc;
    }

    if (lane == 0) {
        int s0 = 0;
#pragma unroll
        for (int e = 1; e < kE; ++e) if (logit[e] > logit[s0]) s0 = e;   // first-max on ties
        int s1 = (s0 == 0) ? 1 : 0;
#pragma unroll
        for (int e = 0; e < kE; ++e) {
            if (e == s0 || e == s1) continue;
            if (logit[e] > logit[s1]) s1 = e;
        }
        // renormalized top-2 softmax weights: w0 = p0/(p0+p1)
        float w0 = __builtin_amdgcn_rcpf(1.0f + __expf(logit[s1] - logit[s0]));
        float w1 = 1.0f - w0;

        int p0 = atomicAdd(&ecnt[s0], 1);
        elist[(size_t)s0 * kT + p0] = tok;
        ew[(size_t)s0 * kT + p0]    = w0;
        int p1 = atomicAdd(&ecnt[s1], 1);
        elist[(size_t)s1 * kT + p1] = tok;
        ew[(size_t)s1 * kT + p1]    = w1;

        zz[tok] = kZ * (w0 * w0 + w1 * w1);
    }
}

// ---------------- finalize: deterministic z_loss reduce + aux/probs ----------
__global__ void moe_finalize_kernel(const float* __restrict__ zz,
                                    const int*   __restrict__ ecnt,
                                    float* __restrict__ aux,
                                    float* __restrict__ zloss,
                                    float* __restrict__ counts,
                                    float* __restrict__ probs)
{
    __shared__ float red[256];
    const int tid = threadIdx.x;
    float s = 0.0f;
    for (int j = tid; j < kT; j += 256) s += zz[j];   // fixed stride order
    red[tid] = s;
    __syncthreads();
    for (int o = 128; o > 0; o >>= 1) {
        if (tid < o) red[tid] += red[tid + o];
        __syncthreads();
    }
    if (tid == 0) {
        *zloss = red[0];
        float a = 0.0f;
#pragma unroll
        for (int e = 0; e < kE; ++e) {
            float c = (float)ecnt[e];
            counts[e] = c;
            float p = c / (float)kT;
            probs[e] = p;
            float d = p - 1.0f / (float)kE;
            a += d * d;
        }
        *aux = kAux * a;
    }
}

// ---------------- sparse grouped GEMMs (bf16 WMMA, f32 accumulate) -----------
// Block: 256 threads = 8 waves, block tile 64(M) x 128(N), wave tile 32x32
// (2x2 of 16x16 WMMA). K-step 32. Early-exit on per-expert row count.
// Template WB: weights already bf16 in workspace (pure u32 LDS copies).
#define LDSS 17   // 16 u32 K-pairs + 1 pad (bank-conflict break)

// GEMM1: inter[pos, :] = silu(x[tok] @ Wg^T) * (x[tok] @ Wu^T), gathered rows.
template <bool WB>
__global__ void __launch_bounds__(256)
moe_gemm1_kernel(const float* __restrict__ x,          // [T,H] fp32
                 const void*  __restrict__ wgp,        // [I,H] expert slice
                 const void*  __restrict__ wup,        // [I,H] expert slice
                 const int*   __restrict__ ecnt_e,     // &ecnt[e]
                 const int*   __restrict__ elist_e,    // &elist[e*T]
                 unsigned short* __restrict__ inter)   // [T,I] bf16 (by list pos)
{
    const int count = *ecnt_e;
    const int tM    = blockIdx.x * 64;
    if (tM >= count) return;
    const int tN    = blockIdx.y * 128;

    __shared__ unsigned int Xs[64 * LDSS];
    __shared__ unsigned int Gs[128 * LDSS];
    __shared__ unsigned int Us[128 * LDSS];
    __shared__ int toks[64];

    const int tid   = threadIdx.x;
    const int wv    = tid >> 5;          // 0..7
    const int lane  = tid & 31;
    const int wm    = wv >> 2;           // 0..1 (M)
    const int wn    = wv & 3;            // 0..3 (N)
    const int laneM = lane & 15;
    const int khalf = lane >> 4;         // 0/1

    if (tid < 64) {
        const int pos = tM + tid;
        toks[tid] = (pos < count) ? elist_e[pos] : elist_e[0];
    }
    __syncthreads();

    v8f accG[4] = {};
    v8f accU[4] = {};

    for (int k0 = 0; k0 < kH; k0 += 32) {
        for (int p = tid; p < 64 * 16; p += 256) {
            const int r = p >> 4, cp = p & 15, gk = k0 + 2 * cp;
            const float* xp = x + (size_t)toks[r] * kH + gk;
            Xs[r * LDSS + cp] = pack2_bf16(xp[0], xp[1]);
        }
        for (int p = tid; p < 128 * 16; p += 256) {
            const int r = p >> 4, cp = p & 15, gk = k0 + 2 * cp;
            if constexpr (WB) {
                const unsigned short* gp = (const unsigned short*)wgp;
                const unsigned short* up = (const unsigned short*)wup;
                Gs[r * LDSS + cp] = *(const unsigned int*)(gp + (size_t)(tN + r) * kH + gk);
                Us[r * LDSS + cp] = *(const unsigned int*)(up + (size_t)(tN + r) * kH + gk);
            } else {
                const float* gp = (const float*)wgp + (size_t)(tN + r) * kH + gk;
                const float* up = (const float*)wup + (size_t)(tN + r) * kH + gk;
                Gs[r * LDSS + cp] = pack2_bf16(gp[0], gp[1]);
                Us[r * LDSS + cp] = pack2_bf16(up[0], up[1]);
            }
        }
        __syncthreads();

        FragU a[2], bg[2], bu[2];
#pragma unroll
        for (int mi = 0; mi < 2; ++mi)
#pragma unroll
            for (int j = 0; j < 8; ++j) {
                // A: lanes 0-15 K {0..7,16..23}; lanes 16-31 K {8..15,24..31}
                const int pa = (j & 3) + ((j & 4) << 1) + (khalf << 2);
                a[mi].u[j] = Xs[(wm * 32 + mi * 16 + laneM) * LDSS + pa];
            }
#pragma unroll
        for (int ni = 0; ni < 2; ++ni)
#pragma unroll
            for (int j = 0; j < 8; ++j) {
                // B: lanes 0-15 K {0..7,16..23}; lanes 16-31 K {8..15,24..31}
                const int pb = (j & 3) + ((j & 4) << 1) + (khalf << 2);
                bg[ni].u[j] = Gs[(wn * 32 + ni * 16 + laneM) * LDSS + pb];
                bu[ni].u[j] = Us[(wn * 32 + ni * 16 + laneM) * LDSS + pb];
            }
#pragma unroll
        for (int mi = 0; mi < 2; ++mi)
#pragma unroll
            for (int ni = 0; ni < 2; ++ni) {
                accG[mi * 2 + ni] = __builtin_amdgcn_wmma_f32_16x16x32_bf16(
                    false, a[mi].v, false, bg[ni].v, (short)0, accG[mi * 2 + ni], false, false);
                accU[mi * 2 + ni] = __builtin_amdgcn_wmma_f32_16x16x32_bf16(
                    false, a[mi].v, false, bu[ni].v, (short)0, accU[mi * 2 + ni], false, false);
            }
        __syncthreads();
    }

    // epilogue: silu(gate)*up -> bf16 inter (indexed by list position)
    const bool full = (tM + 64) <= count;
#pragma unroll
    for (int mi = 0; mi < 2; ++mi) {
#pragma unroll
        for (int ni = 0; ni < 2; ++ni) {
#pragma unroll
            for (int r = 0; r < 8; ++r) {
                const int pos = tM + wm * 32 + mi * 16 + r + khalf * 8;
                if (full || pos < count) {
                    const int col = tN + wn * 32 + ni * 16 + laneM;
                    const float g = accG[mi * 2 + ni][r];
                    const float u = accU[mi * 2 + ni][r];
                    const float s = g * __builtin_amdgcn_rcpf(1.0f + __expf(-g));
                    inter[(size_t)pos * kI + col] = f2bf16(s * u);
                }
            }
        }
    }
}

// GEMM2: out[tok] += w * (inter[pos] @ Wd^T), scattered rows.
template <bool WB>
__global__ void __launch_bounds__(256)
moe_gemm2_kernel(const unsigned short* __restrict__ inter,  // [T,I] bf16 (by pos)
                 const void*  __restrict__ wdp,             // [H,I] expert slice
                 const int*   __restrict__ ecnt_e,
                 const int*   __restrict__ elist_e,
                 const float* __restrict__ ew_e,
                 float* __restrict__ out)                   // [T,H] accumulate
{
    const int count = *ecnt_e;
    const int tM    = blockIdx.x * 64;
    if (tM >= count) return;
    const int tN    = blockIdx.y * 128;

    __shared__ unsigned int As[64 * LDSS];
    __shared__ unsigned int Bs[128 * LDSS];
    __shared__ int   stok[64];
    __shared__ float swt[64];

    const int tid   = threadIdx.x;
    const int wv    = tid >> 5;
    const int lane  = tid & 31;
    const int wm    = wv >> 2;
    const int wn    = wv & 3;
    const int laneM = lane & 15;
    const int khalf = lane >> 4;

    if (tid < 64) {
        const int pos = tM + tid;
        stok[tid] = (pos < count) ? elist_e[pos] : 0;
        swt[tid]  = (pos < count) ? ew_e[pos]    : 0.0f;
    }

    v8f acc[4] = {};

    for (int k0 = 0; k0 < kI; k0 += 32) {
        for (int p = tid; p < 64 * 16; p += 256) {
            const int r = p >> 4, cp = p & 15, gk = k0 + 2 * cp;
            As[r * LDSS + cp] =
                *(const unsigned int*)(inter + (size_t)(tM + r) * kI + gk);
        }
        for (int p = tid; p < 128 * 16; p += 256) {
            const int r = p >> 4, cp = p & 15, gk = k0 + 2 * cp;
            if constexpr (WB) {
                const unsigned short* dp = (const unsigned short*)wdp;
                Bs[r * LDSS + cp] = *(const unsigned int*)(dp + (size_t)(tN + r) * kI + gk);
            } else {
                const float* dp = (const float*)wdp + (size_t)(tN + r) * kI + gk;
                Bs[r * LDSS + cp] = pack2_bf16(dp[0], dp[1]);
            }
        }
        __syncthreads();

        FragU a[2], b[2];
#pragma unroll
        for (int mi = 0; mi < 2; ++mi)
#pragma unroll
            for (int j = 0; j < 8; ++j) {
                const int pa = (j & 3) + ((j & 4) << 1) + (khalf << 2);
                a[mi].u[j] = As[(wm * 32 + mi * 16 + laneM) * LDSS + pa];
            }
#pragma unroll
        for (int ni = 0; ni < 2; ++ni)
#pragma unroll
            for (int j = 0; j < 8; ++j) {
                const int pb = (j & 3) + ((j & 4) << 1) + (khalf << 2);
                b[ni].u[j] = Bs[(wn * 32 + ni * 16 + laneM) * LDSS + pb];
            }
#pragma unroll
        for (int mi = 0; mi < 2; ++mi)
#pragma unroll
            for (int ni = 0; ni < 2; ++ni)
                acc[mi * 2 + ni] = __builtin_amdgcn_wmma_f32_16x16x32_bf16(
                    false, a[mi].v, false, b[ni].v, (short)0, acc[mi * 2 + ni], false, false);
        __syncthreads();
    }

    const bool full = (tM + 64) <= count;
#pragma unroll
    for (int mi = 0; mi < 2; ++mi) {
#pragma unroll
        for (int ni = 0; ni < 2; ++ni) {
#pragma unroll
            for (int r = 0; r < 8; ++r) {
                const int lrow = wm * 32 + mi * 16 + r + khalf * 8;   // 0..63
                const int pos  = tM + lrow;
                if (full || pos < count) {
                    const int col = tN + wn * 32 + ni * 16 + laneM;
                    out[(size_t)stok[lrow] * kH + col] += swt[lrow] * acc[mi * 2 + ni][r];
                }
            }
        }
    }
}

// ---------------- host-side orchestration ----------------
extern "C" void kernel_launch(void* const* d_in, const int* in_sizes, int n_in,
                              void* d_out, int out_size, void* d_ws, size_t ws_size,
                              hipStream_t stream) {
    (void)in_sizes; (void)n_in; (void)out_size;

    const float* x      = (const float*)d_in[0];   // [B,S,H]
    const float* gate_w = (const float*)d_in[1];   // [E,H]
    const float* w_gate = (const float*)d_in[2];   // [E,I,H]
    const float* w_up   = (const float*)d_in[3];   // [E,I,H]
    const float* w_down = (const float*)d_in[4];   // [E,H,I]

    float* out    = (float*)d_out;                 // [T,H]
    float* aux    = out + (size_t)kT * kH;         // [1]
    float* zloss  = aux + 1;                       // [1]
    float* counts = zloss + 1;                     // [E]
    float* probs  = counts + kE;                   // [E]

    // workspace layout (4-byte aligned throughout)
    int*   ecnt  = (int*)d_ws;                     // [E]
    int*   elist = ecnt + kE;                      // [E,T]
    float* ew    = (float*)(elist + (size_t)kE * kT);   // [E,T]
    float* zz    = ew + (size_t)kE * kT;           // [T]
    unsigned short* inter = (unsigned short*)(zz + kT); // [T,I] bf16
    unsigned short* wbf   = inter + (size_t)kT * kI;    // optional: 3*[E,I,H] bf16
    const size_t   wElems = (size_t)kE * kI * kH;       // per weight tensor
    const size_t   needFull =
        (size_t)((char*)(wbf + 3 * wElems) - (char*)d_ws);
    const bool useWB = ws_size >= needFull;

    // 1) zero accumulation targets (d_out poisoned by harness; ws not re-zeroed)
    moe_zero_f_kernel<<<((kT * kH) + 255) / 256, 256, 0, stream>>>(out, kT * kH);
    moe_zero_i_kernel<<<1, 32, 0, stream>>>(ecnt, kE);

    // 2) router: per-expert token lists + weights, per-token z contributions
    moe_router_kernel<<<kT / 8, 256, 0, stream>>>(x, gate_w, ecnt, elist, ew, zz);

    // 3) deterministic z_loss reduce + counts/probs/aux
    moe_finalize_kernel<<<1, 256, 0, stream>>>(zz, ecnt, aux, zloss, counts, probs);

    // 4) optional one-shot weight conversion to bf16 (removes staging VALU work)
    unsigned short* wgb = wbf;
    unsigned short* wub = wbf + wElems;
    unsigned short* wdb = wbf + 2 * wElems;
    if (useWB) {
        const long n2 = (long)(wElems / 2);
        moe_cvt_bf16_kernel<<<4096, 256, 0, stream>>>(w_gate, wgb, n2);
        moe_cvt_bf16_kernel<<<4096, 256, 0, stream>>>(w_up,   wub, n2);
        moe_cvt_bf16_kernel<<<4096, 256, 0, stream>>>(w_down, wdb, n2);
    }

    // 5) sparse grouped expert compute (top-2 only; blocks early-exit on count)
    const dim3 blk(256);
    const dim3 g1(kT / 64, kI / 128);
    const dim3 g2(kT / 64, kH / 128);
    for (int e = 0; e < kE; ++e) {
        const size_t wOff = (size_t)e * kI * kH;   // I*H == H*I
        const int*   cnt  = ecnt + e;
        const int*   lst  = elist + (size_t)e * kT;
        const float* wts  = ew + (size_t)e * kT;
        if (useWB) {
            moe_gemm1_kernel<true><<<g1, blk, 0, stream>>>(
                x, wgb + wOff, wub + wOff, cnt, lst, inter);
            moe_gemm2_kernel<true><<<g2, blk, 0, stream>>>(
                inter, wdb + wOff, cnt, lst, wts, out);
        } else {
            moe_gemm1_kernel<false><<<g1, blk, 0, stream>>>(
                x, w_gate + wOff, w_up + wOff, cnt, lst, inter);
            moe_gemm2_kernel<false><<<g2, blk, 0, stream>>>(
                inter, w_down + wOff, cnt, lst, wts, out);
        }
    }
}
